// TEMoEFFN_37108517437871
// MI455X (gfx1250) — matrix operand order repaired and checked by reading, not compiled, not run on hardware
//
#include <hip/hip_runtime.h>
#include <hip/hip_bf16.h>
#include <math.h>

// Problem constants (from reference): B=2, T=2048 -> S=4096, C=1024, F=512, E=8, top-2
#define S_TOK 4096
#define C_DIM 1024
#define F_DIM 512
#define E_NUM 8

typedef __attribute__((ext_vector_type(16))) __bf16 v16bf;
typedef __attribute__((ext_vector_type(8)))  float  v8f;

union FragAB { v16bf v; uint4 q[2]; };

__device__ __forceinline__ unsigned short f2bf(float f) {
  union { float f; unsigned u; } c; c.f = f;
  unsigned u = c.u + 0x7FFFu + ((c.u >> 16) & 1u);  // round-to-nearest-even
  return (unsigned short)(u >> 16);
}

__device__ __forceinline__ float silu_f(float x) { return x / (1.f + __expf(-x)); }

// ---------------- workspace layout (bytes) ----------------
constexpr size_t OFF_PROBS = 0;
constexpr size_t SZ_PROBS  = (size_t)S_TOK * E_NUM * 4;
constexpr size_t OFF_LISTS = OFF_PROBS + SZ_PROBS;
constexpr size_t SZ_LISTS  = (size_t)E_NUM * S_TOK * 4;
constexpr size_t OFF_COUNTS = OFF_LISTS + SZ_LISTS;
constexpr size_t SZ_COUNTS  = 256;
constexpr size_t OFF_XB    = OFF_COUNTS + SZ_COUNTS;
constexpr size_t SZ_XB     = (size_t)S_TOK * C_DIM * 2;
constexpr size_t OFF_WSH1  = OFF_XB + SZ_XB;
constexpr size_t SZ_WSH1   = (size_t)2 * F_DIM * C_DIM * 2;
constexpr size_t OFF_WSH2  = OFF_WSH1 + SZ_WSH1;
constexpr size_t SZ_WSH2   = (size_t)C_DIM * F_DIM * 2;
constexpr size_t OFF_WF1   = OFF_WSH2 + SZ_WSH2;
constexpr size_t SZ_WF1    = (size_t)E_NUM * 2 * F_DIM * C_DIM * 2;
constexpr size_t OFF_WF2   = OFF_WF1 + SZ_WF1;
constexpr size_t SZ_WF2    = (size_t)E_NUM * C_DIM * F_DIM * 2;
constexpr size_t OFF_ACTS  = OFF_WF2 + SZ_WF2;
constexpr size_t SZ_ACTS   = (size_t)S_TOK * F_DIM * 2;
constexpr size_t OFF_ACTM  = OFF_ACTS + SZ_ACTS;
// total ~74.7 MB

// ---------------- kernels ----------------

__global__ void cvt_f32_bf16(const float* __restrict__ in,
                             unsigned short* __restrict__ out, int n) {
  int i = blockIdx.x * blockDim.x + threadIdx.x;
  int stride = gridDim.x * blockDim.x;
  for (; i < n; i += stride) out[i] = f2bf(in[i]);
}

__global__ void init_misc(int* __restrict__ counts, float* __restrict__ out_tail) {
  if (threadIdx.x < E_NUM) counts[threadIdx.x] = 0;
  if (threadIdx.x == 0) out_tail[0] = 0.f;   // aux scalar output of the reference
}

// one wave (32 lanes) per token; E=8 logits -> softmax -> top-2 renormalized
__global__ void router_topk(const float* __restrict__ x, const float* __restrict__ wr,
                            float* __restrict__ probs, int* __restrict__ counts,
                            int* __restrict__ lists) {
  int wave = threadIdx.x >> 5, lane = threadIdx.x & 31;
  int s = blockIdx.x * 8 + wave;
  if (s >= S_TOK) return;
  const float* xp = x + (size_t)s * C_DIM;
  float acc[E_NUM];
#pragma unroll
  for (int e = 0; e < E_NUM; ++e) acc[e] = 0.f;
  for (int c = lane; c < C_DIM; c += 32) {
    float xv = xp[c];
    const float* w = wr + (size_t)c * E_NUM;
#pragma unroll
    for (int e = 0; e < E_NUM; ++e) acc[e] += xv * w[e];
  }
#pragma unroll
  for (int e = 0; e < E_NUM; ++e)
#pragma unroll
    for (int off = 16; off > 0; off >>= 1)
      acc[e] += __shfl_xor(acc[e], off, 32);
  if (lane == 0) {
    float m = acc[0];
    for (int e = 1; e < E_NUM; ++e) m = fmaxf(m, acc[e]);
    float sum = 0.f, sc[E_NUM];
    for (int e = 0; e < E_NUM; ++e) { sc[e] = __expf(acc[e] - m); sum += sc[e]; }
    float inv = 1.f / sum;
    int i1 = 0; float v1 = -1.f;
    for (int e = 0; e < E_NUM; ++e) { sc[e] *= inv; if (sc[e] > v1) { v1 = sc[e]; i1 = e; } }
    int i2 = -1; float v2 = -1.f;
    for (int e = 0; e < E_NUM; ++e) if (e != i1 && sc[e] > v2) { v2 = sc[e]; i2 = e; }
    float wsum = v1 + v2;
    float p1 = v1 / wsum, p2 = v2 / wsum;
    for (int e = 0; e < E_NUM; ++e) probs[(size_t)s * E_NUM + e] = 0.f;
    probs[(size_t)s * E_NUM + i1] = p1;
    probs[(size_t)s * E_NUM + i2] = p2;
    int pos1 = atomicAdd(&counts[i1], 1); lists[i1 * S_TOK + pos1] = s;
    int pos2 = atomicAdd(&counts[i2], 1); lists[i2 * S_TOK + pos2] = s;
  }
}

// fc1 + SwiGLU: act = silu(x @ Wg^T) * (x @ Wu^T); W1 rows [0,F)=gate, [F,2F)=up.
// Block = 8 waves (2x4) -> 64 rows x 128 act-cols; each wave: 32x32 macro-tile
// (2x2 WMMA tiles, gate+up sharing A) => 8 WMMAs per K-step for 6 fragment loads.
template<bool GATHER>
__global__ void fc1_swiglu(const unsigned short* __restrict__ Xb,
                           const unsigned short* __restrict__ W1,
                           unsigned short* __restrict__ Act,
                           const int* __restrict__ lists,
                           const int* __restrict__ counts) {
  int wave = threadIdx.x >> 5, lane = threadIdx.x & 31;
  int e = GATHER ? blockIdx.z : 0;
  int M = GATHER ? counts[e] : S_TOK;
  int rowbase = blockIdx.x * 64 + (wave >> 2) * 32;
  if (rowbase >= M) return;                      // wave-uniform exit (EXEC stays full)
  int colbase = blockIdx.y * 128 + (wave & 3) * 32;
  const unsigned short* w1 = W1 + (size_t)e * (2 * F_DIM) * C_DIM;
  const int* list = lists + e * S_TOK;
  int g = lane >> 4, ln = lane & 15;

  const unsigned short* aptr[2];
#pragma unroll
  for (int i = 0; i < 2; ++i) {
    int row = rowbase + i * 16 + ln;
    int rr = row < M ? row : M - 1;
    int arow = GATHER ? list[rr] : rr;
    aptr[i] = Xb + (size_t)arow * C_DIM + g * 8;
  }
  const unsigned short* bgp[2];
  const unsigned short* bup[2];
#pragma unroll
  for (int j = 0; j < 2; ++j) {
    bgp[j] = w1 + (size_t)(colbase + j * 16 + ln) * C_DIM + g * 8;
    bup[j] = w1 + (size_t)(F_DIM + colbase + j * 16 + ln) * C_DIM + g * 8;
  }

  v8f accg[2][2] = {};
  v8f accu[2][2] = {};
  for (int kb = 0; kb < C_DIM; kb += 32) {
    FragAB a[2], fg[2], fu[2];
#pragma unroll
    for (int i = 0; i < 2; ++i) {
      a[i].q[0] = *(const uint4*)(aptr[i] + kb);
      a[i].q[1] = *(const uint4*)(aptr[i] + kb + 16);
    }
#pragma unroll
    for (int j = 0; j < 2; ++j) {
      fg[j].q[0] = *(const uint4*)(bgp[j] + kb);
      fg[j].q[1] = *(const uint4*)(bgp[j] + kb + 16);
      fu[j].q[0] = *(const uint4*)(bup[j] + kb);
      fu[j].q[1] = *(const uint4*)(bup[j] + kb + 16);
    }
#pragma unroll
    for (int i = 0; i < 2; ++i)
#pragma unroll
      for (int j = 0; j < 2; ++j) {
        accg[i][j] = __builtin_amdgcn_wmma_f32_16x16x32_bf16(
            false, a[i].v, false, fg[j].v, (short)0, accg[i][j], false, false);
        accu[i][j] = __builtin_amdgcn_wmma_f32_16x16x32_bf16(
            false, a[i].v, false, fu[j].v, (short)0, accu[i][j], false, false);
      }
  }

  unsigned short* actp = Act + (GATHER ? (size_t)e * S_TOK * F_DIM : (size_t)0);
#pragma unroll
  for (int i = 0; i < 2; ++i)
#pragma unroll
    for (int j = 0; j < 2; ++j)
#pragma unroll
      for (int r = 0; r < 8; ++r) {
        int mrow = rowbase + i * 16 + r + 8 * g;
        if (mrow < M)
          actp[(size_t)mrow * F_DIM + colbase + j * 16 + ln] =
              f2bf(silu_f(accg[i][j][r]) * accu[i][j][r]);
      }
}

// fc2: out = act @ W2^T (W2 is (C,F) row-major). Shared: plain f32 store.
// MoE: prob-weighted scatter-add into out at the token's row.
// Same 2x2 register blocking: 4 WMMAs per K-step for 4 fragment loads.
template<bool MOE>
__global__ void fc2_combine(const unsigned short* __restrict__ Act,
                            const unsigned short* __restrict__ W2,
                            float* __restrict__ Out,
                            const float* __restrict__ probs,
                            const int* __restrict__ lists,
                            const int* __restrict__ counts) {
  int wave = threadIdx.x >> 5, lane = threadIdx.x & 31;
  int e = MOE ? blockIdx.z : 0;
  int M = MOE ? counts[e] : S_TOK;
  int rowbase = blockIdx.x * 64 + (wave >> 2) * 32;
  if (rowbase >= M) return;
  int colbase = blockIdx.y * 128 + (wave & 3) * 32;
  const unsigned short* w2  = W2  + (size_t)e * C_DIM * F_DIM;
  const unsigned short* act = Act + (MOE ? (size_t)e * S_TOK * F_DIM : (size_t)0);
  const int* list = lists + e * S_TOK;
  int g = lane >> 4, ln = lane & 15;

  const unsigned short* aptr[2];
#pragma unroll
  for (int i = 0; i < 2; ++i) {
    int row = rowbase + i * 16 + ln;
    int rr = row < M ? row : M - 1;
    aptr[i] = act + (size_t)rr * F_DIM + g * 8;
  }
  const unsigned short* bptr[2];
#pragma unroll
  for (int j = 0; j < 2; ++j)
    bptr[j] = w2 + (size_t)(colbase + j * 16 + ln) * F_DIM + g * 8;

  v8f acc[2][2] = {};
  for (int kb = 0; kb < F_DIM; kb += 32) {
    FragAB a[2], b[2];
#pragma unroll
    for (int i = 0; i < 2; ++i) {
      a[i].q[0] = *(const uint4*)(aptr[i] + kb);
      a[i].q[1] = *(const uint4*)(aptr[i] + kb + 16);
    }
#pragma unroll
    for (int j = 0; j < 2; ++j) {
      b[j].q[0] = *(const uint4*)(bptr[j] + kb);
      b[j].q[1] = *(const uint4*)(bptr[j] + kb + 16);
    }
#pragma unroll
    for (int i = 0; i < 2; ++i)
#pragma unroll
      for (int j = 0; j < 2; ++j)
        acc[i][j] = __builtin_amdgcn_wmma_f32_16x16x32_bf16(
            false, a[i].v, false, b[j].v, (short)0, acc[i][j], false, false);
  }

#pragma unroll
  for (int i = 0; i < 2; ++i)
#pragma unroll
    for (int j = 0; j < 2; ++j)
#pragma unroll
      for (int r = 0; r < 8; ++r) {
        int mrow = rowbase + i * 16 + r + 8 * g;
        if (mrow >= M) continue;
        int col = colbase + j * 16 + ln;
        if (MOE) {
          int tok = list[mrow];
          float p = probs[(size_t)tok * E_NUM + e];
          atomicAdd(&Out[(size_t)tok * C_DIM + col], p * acc[i][j][r]);
        } else {
          Out[(size_t)mrow * C_DIM + col] = acc[i][j][r];
        }
      }
}

// ---------------- launch ----------------
extern "C" void kernel_launch(void* const* d_in, const int* in_sizes, int n_in,
                              void* d_out, int out_size, void* d_ws, size_t ws_size,
                              hipStream_t stream) {
  const float* x        = (const float*)d_in[0];
  const float* w_router = (const float*)d_in[1];
  const float* w_sh1    = (const float*)d_in[2];
  const float* w_sh2    = (const float*)d_in[3];
  const float* w_fc1    = (const float*)d_in[4];
  const float* w_fc2    = (const float*)d_in[5];
  float* out = (float*)d_out;

  char* ws = (char*)d_ws;
  float*          probs  = (float*)(ws + OFF_PROBS);
  int*            lists  = (int*)(ws + OFF_LISTS);
  int*            counts = (int*)(ws + OFF_COUNTS);
  unsigned short* xb     = (unsigned short*)(ws + OFF_XB);
  unsigned short* wsh1b  = (unsigned short*)(ws + OFF_WSH1);
  unsigned short* wsh2b  = (unsigned short*)(ws + OFF_WSH2);
  unsigned short* wf1b   = (unsigned short*)(ws + OFF_WF1);
  unsigned short* wf2b   = (unsigned short*)(ws + OFF_WF2);
  unsigned short* act_s  = (unsigned short*)(ws + OFF_ACTS);
  unsigned short* act_m  = (unsigned short*)(ws + OFF_ACTM);

  auto cvt = [&](const float* src, unsigned short* dst, size_t n) {
    int blocks = (int)((n + 1023) / 1024);
    cvt_f32_bf16<<<blocks, 256, 0, stream>>>(src, dst, (int)n);
  };
  cvt(x,     xb,    (size_t)S_TOK * C_DIM);
  cvt(w_sh1, wsh1b, (size_t)2 * F_DIM * C_DIM);
  cvt(w_sh2, wsh2b, (size_t)C_DIM * F_DIM);
  cvt(w_fc1, wf1b,  (size_t)E_NUM * 2 * F_DIM * C_DIM);
  cvt(w_fc2, wf2b,  (size_t)E_NUM * C_DIM * F_DIM);

  init_misc<<<1, 64, 0, stream>>>(counts, out + (size_t)S_TOK * C_DIM);

  router_topk<<<S_TOK / 8, 256, 0, stream>>>(x, w_router, probs, counts, lists);

  // shared expert
  fc1_swiglu<false><<<dim3(S_TOK / 64, F_DIM / 128), 256, 0, stream>>>(
      xb, wsh1b, act_s, lists, counts);
  fc2_combine<false><<<dim3(S_TOK / 64, C_DIM / 128), 256, 0, stream>>>(
      act_s, wsh2b, out, probs, lists, counts);

  // routed experts (grouped GEMM over per-expert token lists; excess tiles early-exit)
  fc1_swiglu<true><<<dim3(S_TOK / 64, F_DIM / 128, E_NUM), 256, 0, stream>>>(
      xb, wf1b, act_m, lists, counts);
  fc2_combine<true><<<dim3(S_TOK / 64, C_DIM / 128, E_NUM), 256, 0, stream>>>(
      act_m, wf2b, out, probs, lists, counts);
}
